// FusedCombineBmm_45165876084941
// MI455X (gfx1250) — matrix operand order repaired and checked by reading, not compiled, not run on hardware
//
#include <hip/hip_runtime.h>
#include <hip/hip_bf16.h>

// ---------------------------------------------------------------------------
// Fused combine-bmm: out[e,b] = gelu( A[e,b] (2048x1024) x W[e,b] (1024x4096)
//                                     + bias[e,:] ), exact-erf GELU, fp32 I/O.
//
// Phase 1 (prepass, once per element): fp32 -> bf16 convert A, convert+
//   transpose W to [n][k].  192 MiB of bf16 operands == L2 capacity.
// Phase 2: bf16 WMMA GEMM, block tile 128x256, wave tile 64x64 (4x4 frags,
//   1.0 ds_load_b128 per WMMA), double-buffered LDS fed by
//   GLOBAL_LOAD_ASYNC_TO_LDS_B128 (ASYNCcnt), V_WMMA_F32_16X16X32_BF16.
// Fallback (if ws too small): round-1 fused kernel.
// ---------------------------------------------------------------------------

typedef __attribute__((ext_vector_type(16))) __bf16 v16bf;
typedef __attribute__((ext_vector_type(8)))  __bf16 v8bf;
typedef __attribute__((ext_vector_type(8)))  float  v8f;
typedef __attribute__((ext_vector_type(4)))  float  v4f;
typedef __attribute__((ext_vector_type(4)))  int    v4i;

#define DIM_E 8
#define DIM_B 2
#define DIM_M 2048
#define DIM_K 1024
#define DIM_N 4096
#define NUM_EB (DIM_E * DIM_B)

// async-path GEMM tiles
#define GM 128
#define GN 256
#define BK 32
#define LDT 40   // LDS row stride in u16 (32 data + 8 pad = 80 B rows)

// fallback GEMM tiles
#define BM 128
#define BN 128

// ---------------- helpers ----------------

__device__ __forceinline__ unsigned short f2bf(float f) {
    __bf16 h = (__bf16)f;   // RNE conversion (native cvt if available)
    return __builtin_bit_cast(unsigned short, h);
}

__device__ __forceinline__ float gelu_exact(float x) {
    return 0.5f * x * (1.0f + erff(x * 0.70710678118654752440f));
}

#if __has_builtin(__builtin_amdgcn_global_load_async_to_lds_b128)
#define HAVE_ASYNC_LDS 1
#else
#define HAVE_ASYNC_LDS 0
#endif

__device__ __forceinline__ void async_copy16(const void* g, void* l) {
#if HAVE_ASYNC_LDS
    __builtin_amdgcn_global_load_async_to_lds_b128(
        (__attribute__((address_space(1))) v4i*)(v4i*)(void*)g,
        (__attribute__((address_space(3))) v4i*)(v4i*)l,
        0, 0);
#else
    *(uint4*)l = *(const uint4*)g;
#endif
}

__device__ __forceinline__ void wait_async_all() {
#if HAVE_ASYNC_LDS
#if __has_builtin(__builtin_amdgcn_s_wait_asynccnt)
    __builtin_amdgcn_s_wait_asynccnt(0);
#else
    asm volatile("s_wait_asynccnt 0" ::: "memory");
#endif
#endif
}

// ---------------- phase 1a: A fp32 -> bf16, same layout [eb][m][k] ----------

__global__ __launch_bounds__(256)
void convert_A_kernel(const float* __restrict__ A, unsigned short* __restrict__ Abf) {
    size_t i = ((size_t)blockIdx.x * 256 + threadIdx.x) * 8;
    v4f a = __builtin_nontemporal_load((const v4f*)(A + i));
    v4f b = __builtin_nontemporal_load((const v4f*)(A + i + 4));
    uint4 o;
    o.x = (unsigned)f2bf(a.x) | ((unsigned)f2bf(a.y) << 16);
    o.y = (unsigned)f2bf(a.z) | ((unsigned)f2bf(a.w) << 16);
    o.z = (unsigned)f2bf(b.x) | ((unsigned)f2bf(b.y) << 16);
    o.w = (unsigned)f2bf(b.z) | ((unsigned)f2bf(b.w) << 16);
    *(uint4*)(Abf + i) = o;
}

// ------------- phase 1b: W fp32 [eb][k][n] -> bf16 [eb][n][k] ---------------

__global__ __launch_bounds__(256)
void convert_transpose_W(const float* __restrict__ W, unsigned short* __restrict__ Wt) {
    __shared__ __align__(16) unsigned short tile[64 * 72];  // 64x64 tile, pad 8
    const int tid = threadIdx.x;
    const int eb  = blockIdx.z;
    const int k0  = blockIdx.y * 64;
    const int n0  = blockIdx.x * 64;
    const float* __restrict__ Wb = W + (size_t)eb * DIM_K * DIM_N;
    unsigned short* __restrict__ Wtb = Wt + (size_t)eb * DIM_N * DIM_K;

    #pragma unroll
    for (int i = 0; i < 4; ++i) {
        const int id = tid + 256 * i;
        const int kr = id >> 4;          // 0..63 (k row)
        const int nq = id & 15;          // float4 group along n
        v4f g = __builtin_nontemporal_load(
            (const v4f*)(Wb + (size_t)(k0 + kr) * DIM_N + n0 + nq * 4));
        tile[(nq * 4 + 0) * 72 + kr] = f2bf(g.x);
        tile[(nq * 4 + 1) * 72 + kr] = f2bf(g.y);
        tile[(nq * 4 + 2) * 72 + kr] = f2bf(g.z);
        tile[(nq * 4 + 3) * 72 + kr] = f2bf(g.w);
    }
    __syncthreads();
    #pragma unroll
    for (int i = 0; i < 4; ++i) {
        const int id  = tid + 256 * i;
        const int nr  = id >> 4;         // 0..63 (n row)
        const int seg = id & 15;         // 4 k-values = 8 B
        const unsigned* p = (const unsigned*)&tile[nr * 72 + seg * 4];
        uint2 v; v.x = p[0]; v.y = p[1];
        *(uint2*)(Wtb + (size_t)(n0 + nr) * DIM_K + k0 + seg * 4) = v;
    }
}

// ------------- phase 2: bf16 WMMA GEMM + bias + exact GELU ------------------
// block tile 128x256, 8 waves as 2(M) x 4(N), wave tile 64x64 (4x4 fragments)

__global__ __launch_bounds__(256)
void bmm_bf16_bias_gelu(const unsigned short* __restrict__ Abf,  // [eb][M][K] bf16
                        const unsigned short* __restrict__ Wt,   // [eb][N][K] bf16
                        const float* __restrict__ Bias,          // [E][N]
                        float* __restrict__ Out)                 // [eb][M][N]
{
    __shared__ __align__(16) unsigned short ldsA[2][GM * LDT];  // 2 x 10 KB
    __shared__ __align__(16) unsigned short ldsB[2][GN * LDT];  // 2 x 20 KB

    const int tid  = threadIdx.x;
    const int lane = tid & 31;
    const int wid  = tid >> 5;
    const int wm   = wid & 1;       // 64-row slab
    const int wn   = wid >> 1;      // 64-col slab
    const int l15  = lane & 15;
    const int half = lane >> 4;

    const int eb = blockIdx.z;
    const int e  = eb >> 1;
    const int m0 = blockIdx.y * GM;
    const int n0 = blockIdx.x * GN;

    const unsigned short* __restrict__ Ab = Abf + (size_t)eb * DIM_M * DIM_K;
    const unsigned short* __restrict__ Bb = Wt  + (size_t)eb * DIM_N * DIM_K;
    float* __restrict__ Ob = Out + (size_t)eb * DIM_M * DIM_N;

    v8f acc[4][4];
    #pragma unroll
    for (int mt = 0; mt < 4; ++mt)
        #pragma unroll
        for (int nt = 0; nt < 4; ++nt)
            acc[mt][nt] = (v8f){0.f, 0.f, 0.f, 0.f, 0.f, 0.f, 0.f, 0.f};

    // stage fill: each row is 64 B = 4 x 16 B segments
    // A: 128 rows * 4 segs = 512 tasks (2/thread); B: 256 rows * 4 = 1024 (4/thread)
    const int srow = tid >> 2;          // 0..63
    const int sseg = tid & 3;

    auto issue_stage = [&](int k0, int buf) {
        #pragma unroll
        for (int i = 0; i < 2; ++i) {
            const int row = srow + 64 * i;
            async_copy16(Ab + (size_t)(m0 + row) * DIM_K + k0 + sseg * 8,
                         &ldsA[buf][row * LDT + sseg * 8]);
        }
        #pragma unroll
        for (int i = 0; i < 4; ++i) {
            const int row = srow + 64 * i;
            async_copy16(Bb + (size_t)(n0 + row) * DIM_K + k0 + sseg * 8,
                         &ldsB[buf][row * LDT + sseg * 8]);
        }
    };

    issue_stage(0, 0);
    wait_async_all();
    __syncthreads();

    int buf = 0;
    for (int k0 = 0; k0 < DIM_K; k0 += BK) {
        if (k0 + BK < DIM_K) issue_stage(k0 + BK, buf ^ 1);

        // B fragments: lane = column, 16 contiguous bf16 per lane
        v16bf bfrag[4];
        #pragma unroll
        for (int nt = 0; nt < 4; ++nt) {
            const int col = wn * 64 + nt * 16 + l15;
            const int kb  = half * 16;
            union { v16bf v; v8bf h[2]; } u;
            u.h[0] = *(const v8bf*)&ldsB[buf][col * LDT + kb];
            u.h[1] = *(const v8bf*)&ldsB[buf][col * LDT + kb + 8];
            bfrag[nt] = u.v;
        }

        #pragma unroll
        for (int mt = 0; mt < 4; ++mt) {
            const int row = wm * 64 + mt * 16 + l15;
            const int kb  = half * 16;
            union { v16bf v; v8bf h[2]; } u;
            u.h[0] = *(const v8bf*)&ldsA[buf][row * LDT + kb];
            u.h[1] = *(const v8bf*)&ldsA[buf][row * LDT + kb + 8];
            #pragma unroll
            for (int nt = 0; nt < 4; ++nt) {
                acc[mt][nt] = __builtin_amdgcn_wmma_f32_16x16x32_bf16(
                    false, u.v, false, bfrag[nt],
                    (short)0, acc[mt][nt], false, false);
            }
        }

        wait_async_all();   // next-stage LDS writes landed (no-op on last iter)
        __syncthreads();    // all waves done reading this buf / see next buf
        buf ^= 1;
    }

    #pragma unroll
    for (int nt = 0; nt < 4; ++nt) {
        const int col = n0 + wn * 64 + nt * 16 + l15;
        const float bv = Bias[(size_t)e * DIM_N + col];
        #pragma unroll
        for (int mt = 0; mt < 4; ++mt) {
            const int rowbase = m0 + wm * 64 + mt * 16 + half * 8;
            #pragma unroll
            for (int r = 0; r < 8; ++r) {
                float x = acc[mt][nt][r] + bv;
                __builtin_nontemporal_store(gelu_exact(x),
                    &Ob[(size_t)(rowbase + r) * DIM_N + col]);
            }
        }
    }
}

// ------------- fallback: fused convert-in-kernel GEMM -----------------------

__global__ __launch_bounds__(256)
void fused_bmm_bias_gelu(const float* __restrict__ A,
                         const float* __restrict__ W,
                         const float* __restrict__ Bias,
                         float* __restrict__ Out)
{
    __shared__ __align__(16) unsigned short ldsA[BM * BK];
    __shared__ __align__(16) unsigned short ldsB[BN * BK];

    const int tid  = threadIdx.x;
    const int lane = tid & 31;
    const int wid  = tid >> 5;
    const int wm   = wid & 1;
    const int wn   = wid >> 1;
    const int l15  = lane & 15;
    const int half = lane >> 4;

    const int eb = blockIdx.z;
    const int e  = eb >> 1;
    const int m0 = blockIdx.y * BM;
    const int n0 = blockIdx.x * BN;

    const float* __restrict__ Ab = A + (size_t)eb * DIM_M * DIM_K;
    const float* __restrict__ Wb = W + (size_t)eb * DIM_K * DIM_N;
    float* __restrict__ Ob       = Out + (size_t)eb * DIM_M * DIM_N;

    v8f acc[4][2];
    #pragma unroll
    for (int mt = 0; mt < 4; ++mt)
        #pragma unroll
        for (int nt = 0; nt < 2; ++nt)
            acc[mt][nt] = (v8f){0.f, 0.f, 0.f, 0.f, 0.f, 0.f, 0.f, 0.f};

    for (int k0 = 0; k0 < DIM_K; k0 += BK) {
        #pragma unroll
        for (int i = 0; i < 4; ++i) {
            const int id  = tid + 256 * i;
            const int row = id >> 3;
            const int q   = id & 7;
            v4f g = *(const v4f*)(Ab + (size_t)(m0 + row) * DIM_K + k0 + q * 4);
            unsigned* dst = (unsigned*)&ldsA[row * BK + q * 4];
            dst[0] = (unsigned)f2bf(g.x) | ((unsigned)f2bf(g.y) << 16);
            dst[1] = (unsigned)f2bf(g.z) | ((unsigned)f2bf(g.w) << 16);
        }
        #pragma unroll
        for (int i = 0; i < 2; ++i) {
            const int id = tid + 256 * i;
            const int kp = id >> 5;
            const int ng = id & 31;
            const float* src = Wb + (size_t)(k0 + kp * 2) * DIM_N + n0 + ng * 4;
            v4f g0 = *(const v4f*)(src);
            v4f g1 = *(const v4f*)(src + DIM_N);
            #pragma unroll
            for (int j = 0; j < 4; ++j) {
                *(unsigned*)&ldsB[(ng * 4 + j) * BK + kp * 2] =
                    (unsigned)f2bf(g0[j]) | ((unsigned)f2bf(g1[j]) << 16);
            }
        }
        __syncthreads();

        v16bf bfrag[2];
        #pragma unroll
        for (int nt = 0; nt < 2; ++nt) {
            const int col = wn * 32 + nt * 16 + l15;
            const int kb  = half * 16;
            union { v16bf v; v8bf h[2]; } u;
            u.h[0] = *(const v8bf*)&ldsB[col * BK + kb];
            u.h[1] = *(const v8bf*)&ldsB[col * BK + kb + 8];
            bfrag[nt] = u.v;
        }
        #pragma unroll
        for (int mt = 0; mt < 4; ++mt) {
            const int row = wm * 64 + mt * 16 + l15;
            const int kb  = half * 16;
            union { v16bf v; v8bf h[2]; } u;
            u.h[0] = *(const v8bf*)&ldsA[row * BK + kb];
            u.h[1] = *(const v8bf*)&ldsA[row * BK + kb + 8];
            #pragma unroll
            for (int nt = 0; nt < 2; ++nt) {
                acc[mt][nt] = __builtin_amdgcn_wmma_f32_16x16x32_bf16(
                    false, u.v, false, bfrag[nt],
                    (short)0, acc[mt][nt], false, false);
            }
        }
        __syncthreads();
    }

    #pragma unroll
    for (int nt = 0; nt < 2; ++nt) {
        const int col = n0 + wn * 32 + nt * 16 + l15;
        const float bv = Bias[(size_t)e * DIM_N + col];
        #pragma unroll
        for (int mt = 0; mt < 4; ++mt) {
            const int rowbase = m0 + wm * 64 + mt * 16 + half * 8;
            #pragma unroll
            for (int r = 0; r < 8; ++r) {
                float x = acc[mt][nt][r] + bv;
                __builtin_nontemporal_store(gelu_exact(x),
                    &Ob[(size_t)(rowbase + r) * DIM_N + col]);
            }
        }
    }
}

// ---------------------------------------------------------------------------

extern "C" void kernel_launch(void* const* d_in, const int* in_sizes, int n_in,
                              void* d_out, int out_size, void* d_ws, size_t ws_size,
                              hipStream_t stream) {
    const float* A    = (const float*)d_in[0];   // [E,B,M,K]
    const float* W    = (const float*)d_in[1];   // [E,B,K,N]
    const float* Bias = (const float*)d_in[2];   // [E,N]
    float* Out        = (float*)d_out;           // [E,B,M,N]

    const size_t nA = (size_t)NUM_EB * DIM_M * DIM_K;   // 33.5M elems
    const size_t nW = (size_t)NUM_EB * DIM_K * DIM_N;   // 67.1M elems
    const size_t needBytes = (nA + nW) * sizeof(unsigned short);  // 192 MiB

    dim3 block(256);

    if (ws_size >= needBytes) {
        unsigned short* Abf = (unsigned short*)d_ws;
        unsigned short* Wt  = Abf + nA;

        convert_A_kernel<<<dim3((unsigned)(nA / 8 / 256)), block, 0, stream>>>(A, Abf);
        convert_transpose_W<<<dim3(DIM_N / 64, DIM_K / 64, NUM_EB), block, 0, stream>>>(W, Wt);
        dim3 gemmGrid(DIM_N / GN, DIM_M / GM, NUM_EB);  // (16, 16, 16)
        bmm_bf16_bias_gelu<<<gemmGrid, block, 0, stream>>>(Abf, Wt, Bias, Out);
    } else {
        dim3 fbGrid(DIM_N / BN, DIM_M / BM, NUM_EB);    // (32, 16, 16)
        fused_bmm_bias_gelu<<<fbGrid, block, 0, stream>>>(A, W, Bias, Out);
    }
}
